// CTRNNCell_31550829756934
// MI455X (gfx1250) — matrix operand order repaired and checked by reading, not compiled, not run on hardware
//
#include <hip/hip_runtime.h>
#include <hip/hip_bf16.h>

// CTRNN, B=8192, D=N=512, 6 unfolds.
//   X_proj = inputs @ W[:512] + bias          (computed once)
//   per unfold: s = s + 0.1*(tanh(X_proj + s @ W[512:]) - s)
// GEMMs use fp16 hi/lo split (fp16x3) on v_wmma_f32_16x16x32_f16, f32 accum.
//
// Workspace layout (66 MiB total):
//   [0,1)MiB  WT_hi   (512x1024 f16, n-major, k-contiguous)
//   [1,2)     WT_lo
//   [2,10)    Xh      (8192x512 f16 = inputs hi)
//   [10,18)   Xl
//   [18,34)   Xproj   (8192x512 f32)
//   [34,42)   S0h   [42,50) S0l   [50,58) S1h   [58,66) S1l

typedef __attribute__((ext_vector_type(16))) _Float16 v16h;
typedef __attribute__((ext_vector_type(8)))  _Float16 v8h;
typedef __attribute__((ext_vector_type(8)))  float    v8f;

#define CTRNN_B 8192
#define CTRNN_N 512
#define CTRNN_K 1024

// f32 -> (hi f16, lo f16) split, elementwise.
__global__ void k_split_f32(const float* __restrict__ src,
                            _Float16* __restrict__ hi,
                            _Float16* __restrict__ lo, int n) {
  int i = blockIdx.x * blockDim.x + threadIdx.x;
  if (i >= n) return;
  float x = src[i];
  _Float16 h = (_Float16)x;
  hi[i] = h;
  lo[i] = (_Float16)(x - (float)h);
}

// Transpose W (1024x512, row=k) into WT (512x1024, row=n, k-contiguous) + split.
__global__ void k_prep_w(const float* __restrict__ W,
                         _Float16* __restrict__ wth,
                         _Float16* __restrict__ wtl) {
  int i = blockIdx.x * blockDim.x + threadIdx.x;  // 0 .. 512*1024-1
  int n = i >> 10;
  int k = i & 1023;
  float x = W[(size_t)k * CTRNN_N + n];
  _Float16 h = (_Float16)x;
  wth[(size_t)n * CTRNN_K + k] = h;
  wtl[(size_t)n * CTRNN_K + k] = (_Float16)(x - (float)h);
}

// Load one 16-lane-half A fragment: h[0..15] = p[0..15]
// (contiguous K order matching B: lanes<16 hold K 0-15; lanes>=16 hold K 16-31,
//  handled by the caller's +16 base offset.)
__device__ __forceinline__ v16h load_afrag(const _Float16* __restrict__ p) {
  v8h a = *(const v8h*)(p);
  v8h b = *(const v8h*)(p + 8);
  v16h r;
#pragma unroll
  for (int i = 0; i < 8; ++i) { r[i] = a[i]; r[8 + i] = b[i]; }
  return r;
}

// One wave per block, 64x16 output tile, K = 512 in steps of 32.
// mode 0: xproj = A @ WT[:,kOff..kOff+512) + bias          (A = split inputs)
// mode 1: f = tanh(xproj + S @ W_bot);  ns = s + 0.1*(f-s) (A = split state)
//         non-final: write split ns to NSh/NSl; final: write f32 ns to out_f32.
__global__ __launch_bounds__(32) void k_ctrnn_gemm(
    const _Float16* __restrict__ Ah, const _Float16* __restrict__ Al,
    const _Float16* __restrict__ WTh, const _Float16* __restrict__ WTl,
    int kOff,
    const float* __restrict__ bias,
    float* __restrict__ xproj,
    float* __restrict__ out_f32,
    _Float16* __restrict__ NSh, _Float16* __restrict__ NSl,
    int mode) {
  const int lane = threadIdx.x;
  const int row0 = blockIdx.x * 64;
  const int col  = blockIdx.y * 16 + (lane & 15);
  const int half = (lane >> 4) & 1;
  const int aoff = half * 16;  // A-frag lane-half K offset
  // B fragment: N = col, K(base) = kt + half*16, 16 contiguous f16.
  const _Float16* wh = WTh + (size_t)col * CTRNN_K + kOff + half * 16;
  const _Float16* wl = WTl + (size_t)col * CTRNN_K + kOff + half * 16;

  v8f acc0 = {}, acc1 = {}, acc2 = {}, acc3 = {};

  for (int kt = 0; kt < 512; kt += 32) {
    v16h bh = *(const v16h*)(wh + kt);
    v16h bl = *(const v16h*)(wl + kt);
#define CTRNN_SUBTILE(T, ACC)                                                        \
    do {                                                                             \
      const size_t abase = (size_t)(row0 + (T) * 16 + (lane & 15)) * 512 + kt + aoff;\
      v16h ahf = load_afrag(Ah + abase);                                             \
      v16h alf = load_afrag(Al + abase);                                             \
      ACC = __builtin_amdgcn_wmma_f32_16x16x32_f16(false, ahf, false, bh, (short)0, ACC, false, false); \
      ACC = __builtin_amdgcn_wmma_f32_16x16x32_f16(false, ahf, false, bl, (short)0, ACC, false, false); \
      ACC = __builtin_amdgcn_wmma_f32_16x16x32_f16(false, alf, false, bh, (short)0, ACC, false, false); \
    } while (0)
    CTRNN_SUBTILE(0, acc0);
    CTRNN_SUBTILE(1, acc1);
    CTRNN_SUBTILE(2, acc2);
    CTRNN_SUBTILE(3, acc3);
#undef CTRNN_SUBTILE
  }

  // C/D layout: VGPR r <-> M = r + 8*half (per 16x16 subtile), N = col.
  const int mBase = row0 + half * 8;
  if (mode == 0) {
    float bv = bias[col];
#pragma unroll
    for (int r = 0; r < 8; ++r) {
      xproj[(size_t)(mBase +  0 + r) * CTRNN_N + col] = acc0[r] + bv;
      xproj[(size_t)(mBase + 16 + r) * CTRNN_N + col] = acc1[r] + bv;
      xproj[(size_t)(mBase + 32 + r) * CTRNN_N + col] = acc2[r] + bv;
      xproj[(size_t)(mBase + 48 + r) * CTRNN_N + col] = acc3[r] + bv;
    }
  } else {
#pragma unroll
    for (int r = 0; r < 8; ++r) {
      float av[4] = {acc0[r], acc1[r], acc2[r], acc3[r]};
#pragma unroll
      for (int t = 0; t < 4; ++t) {
        const int m = mBase + t * 16 + r;
        const size_t idx = (size_t)m * CTRNN_N + col;
        float s  = (float)Ah[idx] + (float)Al[idx];   // reconstruct current state
        float f  = tanhf(xproj[idx] + av[t]);
        float ns = s + 0.1f * (f - s);                // dt=0.1, tau=1
        if (out_f32) {
          out_f32[idx] = ns;
        } else {
          _Float16 h = (_Float16)ns;
          NSh[idx] = h;
          NSl[idx] = (_Float16)(ns - (float)h);
        }
      }
    }
  }
}

extern "C" void kernel_launch(void* const* d_in, const int* in_sizes, int n_in,
                              void* d_out, int out_size, void* d_ws, size_t ws_size,
                              hipStream_t stream) {
  const float* inputs = (const float*)d_in[0];  // (8192, 512)
  const float* state  = (const float*)d_in[1];  // (8192, 512)
  const float* W      = (const float*)d_in[2];  // (1024, 512)
  const float* bias   = (const float*)d_in[3];  // (512,)
  float* out = (float*)d_out;                   // (state, state) concat

  char* ws = (char*)d_ws;
  const size_t MB = (size_t)1 << 20;
  _Float16* WTh = (_Float16*)(ws + 0 * MB);
  _Float16* WTl = (_Float16*)(ws + 1 * MB);
  _Float16* Xh  = (_Float16*)(ws + 2 * MB);
  _Float16* Xl  = (_Float16*)(ws + 10 * MB);
  float*    XP  = (float*)   (ws + 18 * MB);
  _Float16* S0h = (_Float16*)(ws + 34 * MB);
  _Float16* S0l = (_Float16*)(ws + 42 * MB);
  _Float16* S1h = (_Float16*)(ws + 50 * MB);
  _Float16* S1l = (_Float16*)(ws + 58 * MB);

  const int BN = CTRNN_B * CTRNN_N;  // 4,194,304

  // Prep: transpose+split W; split inputs; split initial state.
  k_prep_w<<<(512 * 1024) / 256, 256, 0, stream>>>(W, WTh, WTl);
  k_split_f32<<<BN / 256, 256, 0, stream>>>(inputs, Xh, Xl, BN);
  k_split_f32<<<BN / 256, 256, 0, stream>>>(state, S0h, S0l, BN);

  dim3 grid(CTRNN_B / 64, CTRNN_N / 16);  // 128 x 32, one wave each

  // X_proj = inputs @ W_top + bias (constant across unfolds).
  k_ctrnn_gemm<<<grid, 32, 0, stream>>>(Xh, Xl, WTh, WTl, /*kOff=*/0,
                                        bias, XP, nullptr, nullptr, nullptr, 0);

  // 6 unfolds, ping-pong split-state buffers; last writes f32 to d_out.
  _Float16 *ph = S0h, *pl = S0l, *qh = S1h, *ql = S1l;
  for (int it = 0; it < 6; ++it) {
    const bool last = (it == 5);
    k_ctrnn_gemm<<<grid, 32, 0, stream>>>(ph, pl, WTh, WTl, /*kOff=*/512,
                                          nullptr, XP,
                                          last ? out : nullptr, qh, ql, 1);
    _Float16* t;
    t = ph; ph = qh; qh = t;
    t = pl; pl = ql; ql = t;
  }

  // Tuple output (state, state): duplicate into second half.
  if (out_size >= 2 * BN) {
    hipMemcpyAsync(out + BN, out, (size_t)BN * sizeof(float),
                   hipMemcpyDeviceToDevice, stream);
  }
}
